// GraphWavenet_63780264345947
// MI455X (gfx1250) — matrix operand order repaired and checked by reading, not compiled, not run on hardware
//
#include <hip/hip_runtime.h>
#include <hip/hip_bf16.h>
#include <cstdint>

typedef __attribute__((ext_vector_type(16))) __bf16 v16bf;
typedef __attribute__((ext_vector_type(8)))  float  v8f;

#define BN    13248
#define EDGES 211968
#define CH    32

// ---------------------------------------------------------------------------
// CDNA5 hardware tanh (V_TANH_F32, TRANS pipe). Sigmoid via exact identity
// sigmoid(x) = 0.5*tanh(x/2) + 0.5  — avoids libm exp expansion entirely.
// ---------------------------------------------------------------------------
__device__ __forceinline__ float fast_tanh(float x) {
#if __has_builtin(__builtin_amdgcn_tanhf)
  return __builtin_amdgcn_tanhf(x);
#else
  float r;
  // TRANS hazard: 1 independent op or V_NOP before result use (ISA §7.4 note)
  asm volatile("v_tanh_f32 %0, %1\n\tv_nop" : "=v"(r) : "v"(x));
  return r;
#endif
}
__device__ __forceinline__ float fast_sigmoid(float x) {
  return fmaf(0.5f, fast_tanh(0.5f * x), 0.5f);
}

__device__ __forceinline__ float bf16lo(uint32_t p) { return __uint_as_float(p << 16); }
__device__ __forceinline__ float bf16hi(uint32_t p) { return __uint_as_float(p & 0xffff0000u); }

// ---------------------------------------------------------------------------
// init linear: (BN*13, 2) @ (2,32) + b  -> h (BN,13,32)   (K=2, no WMMA needed)
// ---------------------------------------------------------------------------
__global__ void init_kernel(const float* __restrict__ x, const float* __restrict__ iw,
                            const float* __restrict__ ib, float* __restrict__ h, int total) {
  int idx = blockIdx.x * blockDim.x + threadIdx.x;
  if (idx >= total) return;
  int r = idx >> 5, c = idx & 31;
  h[idx] = x[r * 2] * iw[c] + x[r * 2 + 1] * iw[32 + c] + ib[c];
}

__global__ void fill_kernel(float* __restrict__ p, float v, int n) {
  int i = blockIdx.x * blockDim.x + threadIdx.x;
  if (i < n) p[i] = v;
}

// ---------------------------------------------------------------------------
// Gated TCN, fused: t = tanh(h0@F0 + hd@F1 + fb) * sigmoid(h0@G0 + hd@G1 + gb)
// One wave -> 16 rows x 32 cols via 8x v_wmma_f32_16x16x32_bf16.
// ---------------------------------------------------------------------------
__global__ __launch_bounds__(256)
void tcn_kernel(const float* __restrict__ h, const float* __restrict__ fw,
                const float* __restrict__ fb, const float* __restrict__ gw,
                const float* __restrict__ gb, float* __restrict__ t,
                int Lin, int Lout, int dil, int ntiles) {
  int wave = threadIdx.x >> 5;
  int lane = threadIdx.x & 31;
  int tile = blockIdx.x * (blockDim.x >> 5) + wave;
  if (tile >= ntiles) return;

  // B fragments: lane&15 = N (nb*16+lane&15); element i = K in table order (lane half selects 8-group).
  v16bf bf0[2], bf1[2], bg0[2], bg1[2];
#pragma unroll
  for (int nb = 0; nb < 2; nb++) {
#pragma unroll
    for (int i = 0; i < 16; i++) {
      int n = nb * 16 + (lane & 15);
      int k = (i < 8 ? 0 : 8) + (lane >> 4) * 8 + i;
      bf0[nb][i] = (__bf16)fw[(n * CH + k) * 2 + 0];
      bf1[nb][i] = (__bf16)fw[(n * CH + k) * 2 + 1];
      bg0[nb][i] = (__bf16)gw[(n * CH + k) * 2 + 0];
      bg1[nb][i] = (__bf16)gw[(n * CH + k) * 2 + 1];
    }
  }

  // A fragments: lane&15 = row M; lane half selects K {0-7,16-23} vs {8-15,24-31}
  int m  = lane & 15;
  int hh = lane >> 4;
  int r  = tile * 16 + m;
  int node = r / Lout, tt = r % Lout;
  const float* row0 = h + ((size_t)node * Lin + tt) * CH;
  const float* row1 = row0 + (size_t)dil * CH;
  v16bf a0, a1;
#pragma unroll
  for (int i = 0; i < 8; i++) {
    a0[i]     = (__bf16)row0[hh * 8 + i];
    a0[8 + i] = (__bf16)row0[16 + hh * 8 + i];
    a1[i]     = (__bf16)row1[hh * 8 + i];
    a1[8 + i] = (__bf16)row1[16 + hh * 8 + i];
  }

  int col = lane & 15;
#pragma unroll
  for (int nb = 0; nb < 2; nb++) {
    v8f cf, cg;
    float fbv = fb[nb * 16 + col], gbv = gb[nb * 16 + col];
#pragma unroll
    for (int i = 0; i < 8; i++) { cf[i] = fbv; cg[i] = gbv; }
    cf = __builtin_amdgcn_wmma_f32_16x16x32_bf16(false, a0, false, bf0[nb], (short)0, cf, false, false);
    cf = __builtin_amdgcn_wmma_f32_16x16x32_bf16(false, a1, false, bf1[nb], (short)0, cf, false, false);
    cg = __builtin_amdgcn_wmma_f32_16x16x32_bf16(false, a0, false, bg0[nb], (short)0, cg, false, false);
    cg = __builtin_amdgcn_wmma_f32_16x16x32_bf16(false, a1, false, bg1[nb], (short)0, cg, false, false);
#pragma unroll
    for (int i = 0; i < 8; i++) {
      int rowm = tile * 16 + hh * 8 + i;
      float v = fast_tanh(cf[i]) * fast_sigmoid(cg[i]);
      t[(size_t)rowm * CH + nb * 16 + col] = v;
    }
  }
}

// ---------------------------------------------------------------------------
// GCN hop transform: Y[:, a] = t @ gcn_w[l,a]   (bf16 output for cheap gathers)
// Y layout: (BN, 6, Lout, 32) bf16.  blockIdx.y = adjacency a (0..5).
// ---------------------------------------------------------------------------
__global__ __launch_bounds__(256)
void gcn_kernel(const float* __restrict__ t, const float* __restrict__ gw,
                __bf16* __restrict__ Y, int Lout, int ntiles) {
  int a    = blockIdx.y;
  int wave = threadIdx.x >> 5;
  int lane = threadIdx.x & 31;
  int tile = blockIdx.x * (blockDim.x >> 5) + wave;
  if (tile >= ntiles) return;

  const float* W = gw + (size_t)a * CH * CH;
  v16bf b[2];
#pragma unroll
  for (int nb = 0; nb < 2; nb++)
#pragma unroll
    for (int i = 0; i < 16; i++) b[nb][i] = (__bf16)W[((i < 8 ? 0 : 8) + (lane >> 4) * 8 + i) * CH + nb * 16 + (lane & 15)];

  int m = lane & 15, hh = lane >> 4;
  const float* row = t + ((size_t)(tile * 16 + m)) * CH;
  v16bf af;
#pragma unroll
  for (int i = 0; i < 8; i++) {
    af[i]     = (__bf16)row[hh * 8 + i];
    af[8 + i] = (__bf16)row[16 + hh * 8 + i];
  }

  int col = lane & 15;
#pragma unroll
  for (int nb = 0; nb < 2; nb++) {
    v8f c;
#pragma unroll
    for (int i = 0; i < 8; i++) c[i] = 0.f;
    c = __builtin_amdgcn_wmma_f32_16x16x32_bf16(false, af, false, b[nb], (short)0, c, false, false);
#pragma unroll
    for (int i = 0; i < 8; i++) {
      int rm = tile * 16 + hh * 8 + i;
      int node = rm / Lout, tt = rm % Lout;
      Y[(((size_t)node * 6 + a) * Lout + tt) * CH + nb * 16 + col] = (__bf16)c[i];
    }
  }
}

// ---------------------------------------------------------------------------
// Edge diffusion: g[to] += w^1*Y[from,k0] + w^2*Y[from,k0+1] + w^3*Y[from,k0+2]
// One wave per (edge, direction); lanes stripe the vector as packed bf16 pairs
// (b32 loads, shift-widened to f32). Y + g fit in the 192MB L2 -> L2-resident.
// ---------------------------------------------------------------------------
__global__ __launch_bounds__(256)
void edge_kernel(const __bf16* __restrict__ Y, const int* __restrict__ esrc,
                 const int* __restrict__ edst, const float* __restrict__ ew,
                 float* __restrict__ g, int nE, int Lout) {
  int wave  = (blockIdx.x * blockDim.x + threadIdx.x) >> 5;
  int lane  = threadIdx.x & 31;
  int nwork = 2 * nE;
  if (wave >= nwork) return;
  int e = wave >> 1, dir = wave & 1;
  int s = esrc[e], d = edst[e];
  float w1 = ew[e], w2 = w1 * w1, w3 = w2 * w1;
  int from = dir ? d : s;
  int to   = dir ? s : d;
  int vec  = Lout * CH;             // elements (always even)
  int vecp = vec >> 1;              // bf16 pairs
  const uint32_t* y0 = (const uint32_t*)(Y + ((size_t)from * 6 + (dir ? 3 : 0)) * vec);
  float* gout = g + (size_t)to * vec;
  for (int ip = lane; ip < vecp; ip += 32) {
    uint32_t p1 = y0[ip], p2 = y0[vecp + ip], p3 = y0[2 * vecp + ip];
    float v0 = w1 * bf16lo(p1) + w2 * bf16lo(p2) + w3 * bf16lo(p3);
    float v1 = w1 * bf16hi(p1) + w2 * bf16hi(p2) + w3 * bf16hi(p3);
    unsafeAtomicAdd(&gout[2 * ip],     v0);
    unsafeAtomicAdd(&gout[2 * ip + 1], v1);
  }
}

// ---------------------------------------------------------------------------
// h_new = g + gcn_b + residual (time-aligned slice of h_prev)
// ---------------------------------------------------------------------------
__global__ void finish_kernel(const float* __restrict__ g, const float* __restrict__ gb,
                              const float* __restrict__ hprev, float* __restrict__ hnew,
                              int Lin, int Lout, int dil, int total) {
  int idx = blockIdx.x * blockDim.x + threadIdx.x;
  if (idx >= total) return;
  int c = idx & 31, r = idx >> 5;
  int node = r / Lout, tt = r % Lout;
  hnew[idx] = g[idx] + gb[c] + hprev[((size_t)node * Lin + tt + dil) * CH + c];
}

// ---------------------------------------------------------------------------
// Skip: skip += h[:, L'-1, :] @ skip_w^T + skip_b   (M=BN, K=32, N=256)
// ---------------------------------------------------------------------------
__global__ __launch_bounds__(256)
void skip_kernel(const float* __restrict__ h, int Lout, const float* __restrict__ sw,
                 const float* __restrict__ sb, float* __restrict__ skip, int ntiles) {
  int nb   = blockIdx.y;
  int wave = threadIdx.x >> 5;
  int lane = threadIdx.x & 31;
  int tile = blockIdx.x * (blockDim.x >> 5) + wave;
  if (tile >= ntiles) return;

  int m = lane & 15, hh = lane >> 4, col = lane & 15;
  const float* row = h + (((size_t)(tile * 16 + m)) * Lout + (Lout - 1)) * CH;
  v16bf af;
#pragma unroll
  for (int i = 0; i < 8; i++) {
    af[i]     = (__bf16)row[hh * 8 + i];
    af[8 + i] = (__bf16)row[16 + hh * 8 + i];
  }
#pragma unroll
  for (int sub = 0; sub < 2; sub++) {
    int n0 = nb * 32 + sub * 16;
    v16bf bf;
#pragma unroll
    for (int i = 0; i < 16; i++) bf[i] = (__bf16)sw[(n0 + col) * CH + (i < 8 ? 0 : 8) + hh * 8 + i];
    v8f c;
    float bv = sb[n0 + col];
#pragma unroll
    for (int i = 0; i < 8; i++) c[i] = bv;
    c = __builtin_amdgcn_wmma_f32_16x16x32_bf16(false, af, false, bf, (short)0, c, false, false);
#pragma unroll
    for (int i = 0; i < 8; i++) {
      int rm = tile * 16 + hh * 8 + i;
      skip[(size_t)rm * 256 + n0 + col] += c[i];
    }
  }
}

// ---------------------------------------------------------------------------
// out0: u0 = relu(relu(skip) @ out0_w + out0_b)   (M=BN, K=256, N=512)
// ---------------------------------------------------------------------------
__global__ __launch_bounds__(256)
void out0_kernel(const float* __restrict__ skip, const float* __restrict__ w,
                 const float* __restrict__ b, float* __restrict__ u0, int ntiles) {
  int nb   = blockIdx.y;
  int wave = threadIdx.x >> 5;
  int lane = threadIdx.x & 31;
  int tile = blockIdx.x * (blockDim.x >> 5) + wave;
  if (tile >= ntiles) return;

  int m = lane & 15, hh = lane >> 4, col = lane & 15;
  const float* row = skip + (size_t)(tile * 16 + m) * 256;

  v8f c0, c1;
  float b0 = b[nb * 32 + col], b1 = b[nb * 32 + 16 + col];
#pragma unroll
  for (int i = 0; i < 8; i++) { c0[i] = b0; c1[i] = b1; }

  for (int kb = 0; kb < 8; kb++) {
    v16bf af;
#pragma unroll
    for (int i = 0; i < 8; i++) {
      af[i]     = (__bf16)fmaxf(row[kb * 32 + hh * 8 + i], 0.f);
      af[8 + i] = (__bf16)fmaxf(row[kb * 32 + 16 + hh * 8 + i], 0.f);
    }
    const float* wr = w + (size_t)(kb * 32) * 512 + nb * 32 + col;
    v16bf bf0, bf1;
#pragma unroll
    for (int i = 0; i < 16; i++) { int k = (i < 8 ? 0 : 8) + hh * 8 + i; bf0[i] = (__bf16)wr[(size_t)k * 512]; bf1[i] = (__bf16)wr[(size_t)k * 512 + 16]; }
    c0 = __builtin_amdgcn_wmma_f32_16x16x32_bf16(false, af, false, bf0, (short)0, c0, false, false);
    c1 = __builtin_amdgcn_wmma_f32_16x16x32_bf16(false, af, false, bf1, (short)0, c1, false, false);
  }
#pragma unroll
  for (int i = 0; i < 8; i++) {
    int rm = tile * 16 + hh * 8 + i;
    u0[(size_t)rm * 512 + nb * 32 + col]      = fmaxf(c0[i], 0.f);
    u0[(size_t)rm * 512 + nb * 32 + 16 + col] = fmaxf(c1[i], 0.f);
  }
}

// ---------------------------------------------------------------------------
// out1: out = u0 @ out1_w + out1_b   (K=512, N=12 — too skinny for WMMA)
// ---------------------------------------------------------------------------
__global__ void out1_kernel(const float* __restrict__ u0, const float* __restrict__ w,
                            const float* __restrict__ b, float* __restrict__ out, int Mrows) {
  int tid = blockIdx.x * blockDim.x + threadIdx.x;
  int m = tid >> 4, n = tid & 15;
  if (m >= Mrows || n >= 12) return;
  float acc = b[n];
  const float* row = u0 + (size_t)m * 512;
  for (int k = 0; k < 512; k++) acc += row[k] * w[k * 12 + n];
  out[(size_t)m * 12 + n] = acc;
}

// ---------------------------------------------------------------------------
extern "C" void kernel_launch(void* const* d_in, const int* in_sizes, int n_in,
                              void* d_out, int out_size, void* d_ws, size_t ws_size,
                              hipStream_t stream) {
  const float* x    = (const float*)d_in[0];
  const int*   esrc = (const int*)d_in[1];
  const int*   edst = (const int*)d_in[2];
  const float* ew   = (const float*)d_in[3];
  const float* iw   = (const float*)d_in[4];
  const float* ib   = (const float*)d_in[5];
  const float* fw   = (const float*)d_in[6];
  const float* fb   = (const float*)d_in[7];
  const float* gw   = (const float*)d_in[8];
  const float* gb   = (const float*)d_in[9];
  const float* cw   = (const float*)d_in[10];
  const float* cb   = (const float*)d_in[11];
  const float* sw   = (const float*)d_in[12];
  const float* sb   = (const float*)d_in[13];
  const float* o0w  = (const float*)d_in[14];
  const float* o0b  = (const float*)d_in[15];
  const float* o1w  = (const float*)d_in[16];
  const float* o1b  = (const float*)d_in[17];
  float* out = (float*)d_out;

  char* ws = (char*)d_ws;
  size_t off = 0;
  auto alloc = [&](size_t bytes) -> void* {
    void* p = (void*)(ws + off);
    off += (bytes + 255) & ~(size_t)255;
    return p;
  };
  float*  hA   = (float*)alloc((size_t)BN * 13 * CH * 4);
  float*  hB   = (float*)alloc((size_t)BN * 13 * CH * 4);
  float*  tb   = (float*)alloc((size_t)BN * 12 * CH * 4);
  __bf16* Y    = (__bf16*)alloc((size_t)BN * 6 * 12 * CH * 2);
  float*  g    = (float*)alloc((size_t)BN * 12 * CH * 4);
  float*  skip = (float*)alloc((size_t)BN * 256 * 4);
  float*  u0   = (float*)alloc((size_t)BN * 512 * 4);

  dim3 blk(256);

  {
    int total = BN * 13 * CH;
    init_kernel<<<dim3((total + 255) / 256), blk, 0, stream>>>(x, iw, ib, hA, total);
  }
  fill_kernel<<<dim3((BN * 256 + 255) / 256), blk, 0, stream>>>(skip, 0.f, BN * 256);

  const int dil[8] = {1, 2, 1, 2, 1, 2, 1, 2};
  float* hcur = hA;
  float* hnext = hB;
  int Lin = 13;
  for (int l = 0; l < 8; l++) {
    int d = dil[l];
    int Lout = Lin - d;
    int M = BN * Lout;            // always divisible by 16 (BN = 16*828)
    int ntiles = M / 16;

    tcn_kernel<<<dim3((ntiles + 7) / 8), blk, 0, stream>>>(
        hcur, fw + (size_t)l * CH * CH * 2, fb + l * CH,
        gw + (size_t)l * CH * CH * 2, gb + l * CH, tb, Lin, Lout, d, ntiles);

    gcn_kernel<<<dim3((ntiles + 7) / 8, 6), blk, 0, stream>>>(
        tb, cw + (size_t)l * 6 * CH * CH, Y, Lout, ntiles);

    fill_kernel<<<dim3((M * CH + 255) / 256), blk, 0, stream>>>(g, 0.f, M * CH);

    edge_kernel<<<dim3((2 * EDGES + 7) / 8), blk, 0, stream>>>(
        Y, esrc, edst, ew, g, EDGES, Lout);

    finish_kernel<<<dim3((M * CH + 255) / 256), blk, 0, stream>>>(
        g, cb + l * CH, hcur, hnext, Lin, Lout, d, M * CH);

    skip_kernel<<<dim3((BN / 16 + 7) / 8, 8), blk, 0, stream>>>(
        hnext, Lout, sw + (size_t)l * 256 * CH, sb + (size_t)l * 256, skip, BN / 16);

    float* tmp = hcur; hcur = hnext; hnext = tmp;
    Lin = Lout;
  }

  out0_kernel<<<dim3((BN / 16 + 7) / 8, 16), blk, 0, stream>>>(skip, o0w, o0b, u0, BN / 16);
  out1_kernel<<<dim3((BN * 16 + 255) / 256), blk, 0, stream>>>(u0, o1w, o1b, out, BN);
}
